// Nat3D_11519102288429
// MI455X (gfx1250) — matrix operand order repaired and checked by reading, compiled, not run on hardware
//
#include <hip/hip_runtime.h>
#include <hip/hip_bf16.h>
#include <hip/hip_fp16.h>

typedef __attribute__((ext_vector_type(16))) _Float16 v16h;
typedef __attribute__((ext_vector_type(8)))  _Float16 v8h;
typedef __attribute__((ext_vector_type(8)))  float    v8f;

namespace {
constexpr int D_ = 4, H_ = 48, W_ = 48, C_ = 256, NH_ = 8, HD_ = 32;
constexpr int TOK = D_ * H_ * W_;          // 9216 tokens
constexpr int KD_ = 3, KH_ = 5, KW_ = 5;   // neighborhood 3x5x5 (KVOL=75)
constexpr float SCALE_ = 0.17677669529663687f; // 32^-0.5
}

// ---------------- prep kernels ----------------

__global__ void cvt_f32_to_f16(const float* __restrict__ src,
                               _Float16* __restrict__ dst, int n) {
    int i = blockIdx.x * blockDim.x + threadIdx.x;
    if (i < n) dst[i] = (_Float16)src[i];
}

// dst[n*rows + k] = (f16) src[k*cols + n]   (transpose KxN -> NxK, fp16)
__global__ void transpose_to_f16(const float* __restrict__ src,
                                 _Float16* __restrict__ dst,
                                 int rows, int cols) {
    int i = blockIdx.x * blockDim.x + threadIdx.x;
    if (i < rows * cols) {
        int n = i / rows;
        int k = i - n * rows;
        dst[i] = (_Float16)src[(size_t)k * cols + n];
    }
}

// ---------------- QKV GEMM (WMMA) ----------------
// C[9216][768] = Xh[9216][256] @ Wqkv[256][768] + b; scatter into q/k/v
// (head-major fp16, q scaled). Wave tile: 16(M) x 64(N), K unrolled 8x32.
__global__ __launch_bounds__(128)
void qkv_gemm_wmma(const _Float16* __restrict__ Xh,
                   const _Float16* __restrict__ WT,   // [768][256] (N-major)
                   const float* __restrict__ bqkv,    // [768]
                   _Float16* __restrict__ qh,
                   _Float16* __restrict__ kh,
                   _Float16* __restrict__ vh) {
    int wave = (blockIdx.x * blockDim.x + threadIdx.x) >> 5;
    int lane = threadIdx.x & 31;
    int tileM = wave / 12;            // 576 M-tiles
    int ng    = wave % 12;            // 12 N-groups of 64
    int m0 = tileM * 16;
    int n0 = ng * 64;
    int col  = lane & 15;
    int half = lane >> 4;

    const _Float16* Arow = Xh + (size_t)(m0 + col) * C_;
    const _Float16* Bcol = WT + (size_t)(n0 + col) * C_;

    v8f acc[4] = {};
#pragma unroll
    for (int k0 = 0; k0 < 256; k0 += 32) {
        int kA = k0 + half * 16;
        v8h alo = *(const v8h*)(Arow + kA);
        v8h ahi = *(const v8h*)(Arow + kA + 8);
        v16h a = __builtin_shufflevector(alo, ahi,
                 0,1,2,3,4,5,6,7,8,9,10,11,12,13,14,15);
        int kB = k0 + half * 16;
        v16h b0 = *(const v16h*)(Bcol + 0  * C_ + kB);
        v16h b1 = *(const v16h*)(Bcol + 16 * C_ + kB);
        v16h b2 = *(const v16h*)(Bcol + 32 * C_ + kB);
        v16h b3 = *(const v16h*)(Bcol + 48 * C_ + kB);
        acc[0] = __builtin_amdgcn_wmma_f32_16x16x32_f16(false, a, false, b0,
                                                        (short)0, acc[0], false, false);
        acc[1] = __builtin_amdgcn_wmma_f32_16x16x32_f16(false, a, false, b1,
                                                        (short)0, acc[1], false, false);
        acc[2] = __builtin_amdgcn_wmma_f32_16x16x32_f16(false, a, false, b2,
                                                        (short)0, acc[2], false, false);
        acc[3] = __builtin_amdgcn_wmma_f32_16x16x32_f16(false, a, false, b3,
                                                        (short)0, acc[3], false, false);
    }

#pragma unroll
    for (int t = 0; t < 4; ++t) {
        int n    = n0 + t * 16 + col;
        int part = n >> 8;               // 0=q 1=k 2=v (tile-uniform)
        int head = (n & 255) >> 5;
        int hd   = n & 31;
        float bias = bqkv[n];
        _Float16* dst = (part == 0) ? qh : (part == 1) ? kh : vh;
        float scl = (part == 0) ? SCALE_ : 1.0f;
        size_t base = ((size_t)head * TOK) * HD_ + hd;
#pragma unroll
        for (int r = 0; r < 8; ++r) {
            int m = m0 + r + half * 8;
            dst[base + (size_t)m * HD_] = (_Float16)((acc[t][r] + bias) * scl);
        }
    }
}

// ---------------- neighborhood attention (online softmax) ----------------
// one thread = one (head, token); single pass over 75 neighbors.
__global__ __launch_bounds__(256)
void natten3d_kernel(const _Float16* __restrict__ qh,
                     const _Float16* __restrict__ kh,
                     const _Float16* __restrict__ vh,
                     const float* __restrict__ rpb,     // [8][5][9][9]
                     _Float16* __restrict__ attnh) {    // [token][256]
    int gid = blockIdx.x * blockDim.x + threadIdx.x;
    if (gid >= NH_ * TOK) return;
    int head = gid / TOK;
    int tok  = gid - head * TOK;
    int z = tok / (H_ * W_);
    int rem = tok - z * (H_ * W_);
    int y = rem / W_;
    int x = rem - y * W_;

    int sD = min(max(z - 1, 0), D_ - KD_);
    int sH = min(max(y - 2, 0), H_ - KH_);
    int sW = min(max(x - 2, 0), W_ - KW_);

    const _Float16* qbase = qh + ((size_t)head * TOK + tok) * HD_;
    float q[HD_];
#pragma unroll
    for (int c = 0; c < 4; ++c) {
        v8h qc = *(const v8h*)(qbase + c * 8);
#pragma unroll
        for (int j = 0; j < 8; ++j) q[c * 8 + j] = (float)qc[j];
    }

    const float* rb = rpb + head * (5 * 9 * 9);

    float mval = -3.0e38f, ssum = 0.0f;
    float o[HD_];
#pragma unroll
    for (int c = 0; c < HD_; ++c) o[c] = 0.0f;

    const size_t hbase = (size_t)head * TOK * HD_;
    for (int t = 0; t < KD_; ++t) {
        int nd = sD + t;
        for (int u = 0; u < KH_; ++u) {
            int nh = sH + u;
#pragma unroll
            for (int w = 0; w < KW_; ++w) {
                int nw = sW + w;
                int ntok = (nd * H_ + nh) * W_ + nw;
                const _Float16* kp = kh + hbase + (size_t)ntok * HD_;
                float dot = 0.0f;
#pragma unroll
                for (int c = 0; c < 4; ++c) {
                    v8h kc = *(const v8h*)(kp + c * 8);
#pragma unroll
                    for (int j = 0; j < 8; ++j)
                        dot = fmaf((float)kc[j], q[c * 8 + j], dot);
                }
                float sc = dot + rb[(nd - z + 2) * 81 + (nh - y + 4) * 9 + (nw - x + 4)];
                float nm = fmaxf(mval, sc);
                float corr = __expf(mval - nm);
                float p    = __expf(sc - nm);
                ssum = ssum * corr + p;
                const _Float16* vp = vh + hbase + (size_t)ntok * HD_;
#pragma unroll
                for (int c = 0; c < 4; ++c) {
                    v8h vc = *(const v8h*)(vp + c * 8);
#pragma unroll
                    for (int j = 0; j < 8; ++j)
                        o[c * 8 + j] = fmaf(o[c * 8 + j], corr, p * (float)vc[j]);
                }
                mval = nm;
            }
        }
    }
    float inv = 1.0f / ssum;
    _Float16* dst = attnh + (size_t)tok * C_ + head * HD_;
#pragma unroll
    for (int c = 0; c < HD_; ++c) dst[c] = (_Float16)(o[c] * inv);
}

// ---------------- projection GEMM (WMMA) ----------------
// out[9216][256] = attnh[9216][256] @ Wproj[256][256] + b_proj (fp32 out)
__global__ __launch_bounds__(128)
void proj_gemm_wmma(const _Float16* __restrict__ Ah,
                    const _Float16* __restrict__ WT,   // [256][256] (N-major)
                    const float* __restrict__ bproj,
                    float* __restrict__ out) {
    int wave = (blockIdx.x * blockDim.x + threadIdx.x) >> 5;
    int lane = threadIdx.x & 31;
    int tileM = wave / 4;             // 576 M-tiles
    int ng    = wave % 4;             // 4 N-groups of 64
    int m0 = tileM * 16;
    int n0 = ng * 64;
    int col  = lane & 15;
    int half = lane >> 4;

    const _Float16* Arow = Ah + (size_t)(m0 + col) * C_;
    const _Float16* Bcol = WT + (size_t)(n0 + col) * C_;

    v8f acc[4] = {};
#pragma unroll
    for (int k0 = 0; k0 < 256; k0 += 32) {
        int kA = k0 + half * 16;
        v8h alo = *(const v8h*)(Arow + kA);
        v8h ahi = *(const v8h*)(Arow + kA + 8);
        v16h a = __builtin_shufflevector(alo, ahi,
                 0,1,2,3,4,5,6,7,8,9,10,11,12,13,14,15);
        int kB = k0 + half * 16;
        v16h b0 = *(const v16h*)(Bcol + 0  * C_ + kB);
        v16h b1 = *(const v16h*)(Bcol + 16 * C_ + kB);
        v16h b2 = *(const v16h*)(Bcol + 32 * C_ + kB);
        v16h b3 = *(const v16h*)(Bcol + 48 * C_ + kB);
        acc[0] = __builtin_amdgcn_wmma_f32_16x16x32_f16(false, a, false, b0,
                                                        (short)0, acc[0], false, false);
        acc[1] = __builtin_amdgcn_wmma_f32_16x16x32_f16(false, a, false, b1,
                                                        (short)0, acc[1], false, false);
        acc[2] = __builtin_amdgcn_wmma_f32_16x16x32_f16(false, a, false, b2,
                                                        (short)0, acc[2], false, false);
        acc[3] = __builtin_amdgcn_wmma_f32_16x16x32_f16(false, a, false, b3,
                                                        (short)0, acc[3], false, false);
    }

#pragma unroll
    for (int t = 0; t < 4; ++t) {
        int n = n0 + t * 16 + col;
        float bias = bproj[n];
#pragma unroll
        for (int r = 0; r < 8; ++r) {
            int m = m0 + r + half * 8;
            out[(size_t)m * C_ + n] = acc[t][r] + bias;
        }
    }
}

// ---------------- launch ----------------

extern "C" void kernel_launch(void* const* d_in, const int* in_sizes, int n_in,
                              void* d_out, int out_size, void* d_ws, size_t ws_size,
                              hipStream_t stream) {
    (void)in_sizes; (void)n_in; (void)out_size; (void)ws_size;
    const float* x      = (const float*)d_in[0];   // [9216][256]
    const float* w_qkv  = (const float*)d_in[1];   // [256][768]
    const float* b_qkv  = (const float*)d_in[2];   // [768]
    const float* rpb    = (const float*)d_in[3];   // [8][5][9][9]
    const float* w_proj = (const float*)d_in[4];   // [256][256]
    const float* b_proj = (const float*)d_in[5];   // [256]
    float* out = (float*)d_out;

    char* p = (char*)d_ws;
    _Float16* Xh     = (_Float16*)p; p += (size_t)TOK * C_ * 2;       // 4.72 MB
    _Float16* WqkvT  = (_Float16*)p; p += (size_t)768 * 256 * 2;      // 0.39 MB
    _Float16* WprojT = (_Float16*)p; p += (size_t)256 * 256 * 2;      // 0.13 MB
    _Float16* qh     = (_Float16*)p; p += (size_t)NH_ * TOK * HD_ * 2;
    _Float16* kh     = (_Float16*)p; p += (size_t)NH_ * TOK * HD_ * 2;
    _Float16* vh     = (_Float16*)p; p += (size_t)NH_ * TOK * HD_ * 2;
    _Float16* attnh  = (_Float16*)p; p += (size_t)TOK * C_ * 2;       // total ~23 MB

    // prep: fp16 activation copy + N-major fp16 weight copies
    {
        int n = TOK * C_;
        cvt_f32_to_f16<<<(n + 255) / 256, 256, 0, stream>>>(x, Xh, n);
    }
    {
        int n = 256 * 768;
        transpose_to_f16<<<(n + 255) / 256, 256, 0, stream>>>(w_qkv, WqkvT, 256, 768);
    }
    {
        int n = 256 * 256;
        transpose_to_f16<<<(n + 255) / 256, 256, 0, stream>>>(w_proj, WprojT, 256, 256);
    }

    // QKV GEMM: 576 M-tiles x 12 N-groups = 6912 waves, 4 waves/block
    qkv_gemm_wmma<<<6912 / 4, 128, 0, stream>>>(Xh, WqkvT, b_qkv, qh, kh, vh);

    // neighborhood attention: 8*9216 = 73728 threads
    natten3d_kernel<<<(NH_ * TOK + 255) / 256, 256, 0, stream>>>(qh, kh, vh, rpb, attnh);

    // projection GEMM: 576 x 4 = 2304 waves
    proj_gemm_wmma<<<2304 / 4, 128, 0, stream>>>(attnh, WprojT, b_proj, out);
}
